// MoleculeEncoder_16295105921252
// MI455X (gfx1250) — matrix operand order repaired and checked by reading, not compiled, not run on hardware
//
#include <hip/hip_runtime.h>
#include <hip/hip_bf16.h>

typedef __attribute__((ext_vector_type(16))) __bf16 v16bf;
typedef __attribute__((ext_vector_type(8)))  __bf16 v8bf;
typedef __attribute__((ext_vector_type(8)))  float  v8f;

#define EMB 300
#define LAYERS 5

__device__ __forceinline__ __bf16 to_bf16(float f) {
  union { float f; unsigned u; } c; c.f = f;
  unsigned r = (c.u + 0x7FFFu + ((c.u >> 16) & 1u)) >> 16;
  unsigned short s = (unsigned short)r;
  union { unsigned short s; __bf16 b; } o; o.s = s;
  return o.b;
}

// ---------------------------------------------------------------------------
// Weight pre-swizzle into WMMA B-fragment order:
// Wf[l][kt][nt][lane][i] = W[l][kt*32 + (lane<16?0:8) + (i<8 ? i : 16+i-8)][nt*16 + (lane&15)]
// so each lane's 16-element (32B) B fragment is contiguous in memory.
// ---------------------------------------------------------------------------
__global__ void prep_w_frag(const float* __restrict__ W, __bf16* __restrict__ Wf,
                            int K, int Ncols, int KT, int NT) {
  long idx = (long)blockIdx.x * blockDim.x + threadIdx.x;
  long total = (long)LAYERS * KT * NT * 512;
  if (idx >= total) return;
  int i    = (int)(idx & 15);
  int lane = (int)((idx >> 4) & 31);
  long tile = idx >> 9;
  int nt = (int)(tile % NT);
  long t2 = tile / NT;
  int kt = (int)(t2 % KT);
  int l  = (int)(t2 / KT);
  int k = kt * 32 + ((lane < 16) ? 0 : 8) + ((i < 8) ? i : 16 + (i - 8));
  int n = nt * 16 + (lane & 15);
  float v = 0.f;
  if (k < K && n < Ncols) v = W[(size_t)l * K * Ncols + (size_t)k * Ncols + n];
  Wf[idx] = to_bf16(v);
}

// ---------------------------------------------------------------------------
// Encoders: sum of per-feature embedding rows.
// ---------------------------------------------------------------------------
__global__ void atom_encode(const int* __restrict__ x, const float* __restrict__ emb,
                            float* __restrict__ h, int N) {
  const int aoff[9] = {0, 119, 123, 135, 147, 157, 163, 169, 171};
  long idx = (long)blockIdx.x * blockDim.x + threadIdx.x;
  if (idx >= (long)N * EMB) return;
  int n = (int)(idx / EMB);
  int d = (int)(idx % EMB);
  const int* xr = x + (size_t)n * 9;
  float s = 0.f;
#pragma unroll
  for (int f = 0; f < 9; ++f)
    s += emb[(size_t)(xr[f] + aoff[f]) * EMB + d];
  h[idx] = s;
}

__global__ void bond_encode(const int* __restrict__ eat, const float* __restrict__ emb,
                            float* __restrict__ ea, int E) {
  const int boff[3] = {0, 5, 11};
  long idx = (long)blockIdx.x * blockDim.x + threadIdx.x;
  if (idx >= (long)E * EMB) return;
  int e = (int)(idx / EMB);
  int d = (int)(idx % EMB);
  const int* er = eat + (size_t)e * 3;
  float s = 0.f;
#pragma unroll
  for (int f = 0; f < 3; ++f)
    s += emb[(size_t)(er[f] + boff[f]) * EMB + d];
  ea[idx] = s;
}

// ---------------------------------------------------------------------------
// Edge message: agg[dst] += relu(h[src] + ea), float4-wide.
// ---------------------------------------------------------------------------
__global__ void edge_msg(const float* __restrict__ h, const float* __restrict__ ea,
                         const int* __restrict__ src, const int* __restrict__ dst,
                         float* __restrict__ agg, int E) {
  const int C4 = EMB / 4;  // 75
  long idx = (long)blockIdx.x * blockDim.x + threadIdx.x;
  if (idx >= (long)E * C4) return;
  int e = (int)(idx / C4);
  int c = (int)(idx % C4);
  int s = src[e], d = dst[e];
  float4 hv = ((const float4*)(h + (size_t)s * EMB))[c];
  float4 ev = ((const float4*)(ea + (size_t)e * EMB))[c];
  float m0 = fmaxf(hv.x + ev.x, 0.f);
  float m1 = fmaxf(hv.y + ev.y, 0.f);
  float m2 = fmaxf(hv.z + ev.z, 0.f);
  float m3 = fmaxf(hv.w + ev.w, 0.f);
  float* ap = agg + (size_t)d * EMB + (size_t)c * 4;
  atomicAdd(ap + 0, m0);
  atomicAdd(ap + 1, m1);
  atomicAdd(ap + 2, m2);
  atomicAdd(ap + 3, m3);
}

// z = h + agg  ->  bf16 padded [N,320] GEMM1 operand
__global__ void make_a1(const float* __restrict__ h, const float* __restrict__ agg,
                        __bf16* __restrict__ a1, int N) {
  long idx = (long)blockIdx.x * blockDim.x + threadIdx.x;
  if (idx >= (long)N * 320) return;
  int n = (int)(idx / 320);
  int c = (int)(idx % 320);
  float z = 0.f;
  if (c < EMB) z = h[(size_t)n * EMB + c] + agg[(size_t)n * EMB + c];
  a1[idx] = to_bf16(z);
}

// ---------------------------------------------------------------------------
// bf16 WMMA GEMM: C[M, ncols] = A[M, lda(bf16)] @ Bfrag + bias.
// One wave (wave32) per 16x32 C block (2 adjacent 16x16 N-tiles), so each
// A fragment feeds 2 WMMAs. 8 waves/block; K in steps of 32; f32 accumulate.
// A row-major padded bf16; B pre-swizzled fragments (adjacent 1KB blocks per
// n-tile) walked with incremented pointers so loads use immediate offsets.
// ---------------------------------------------------------------------------
__global__ void __launch_bounds__(256) gemm_bf16_wmma(
    const __bf16* __restrict__ A, int lda,
    const __bf16* __restrict__ Bf,
    const float* __restrict__ bias,
    float* __restrict__ C, int ldc,
    int M, int KT, int NT, int ncols) {
  const int lane = threadIdx.x & 31;
  const int wave = threadIdx.x >> 5;
  const int NG = NT >> 1;                       // n-groups of 2 tiles
  const long grp = (long)blockIdx.x * 8 + wave;
  const int MT = (M + 15) >> 4;
  if (grp >= (long)MT * NG) return;
  const int mt = (int)(grp / NG);
  const int ng = (int)(grp % NG);
  const int nt0 = ng * 2;

  int rowA = mt * 16 + (lane & 15);
  if (rowA >= M) rowA = M - 1;  // clamp; stores are row-guarded
  const int k0 = (lane < 16) ? 0 : 8;           // ISA 16-bit A-fragment K split
  const __bf16* __restrict__ aptr = A + (size_t)rowA * lda + k0;
  const __bf16* __restrict__ bptr =
      Bf + ((size_t)nt0 << 9) + ((size_t)lane << 4);
  const size_t bstride = (size_t)NT << 9;       // fragment elems per k-step

  v8f acc0 = {};
  v8f acc1 = {};
  for (int kt = 0; kt < KT; ++kt) {
    v8bf alo = *(const v8bf*)(aptr);            // K +0..7 (or +8..15)
    v8bf ahi = *(const v8bf*)(aptr + 16);       // K +16..23 (or +24..31)
    v16bf av;
#pragma unroll
    for (int i = 0; i < 8; ++i) { av[i] = alo[i]; av[i + 8] = ahi[i]; }
    v16bf bv0 = *(const v16bf*)(bptr);
    v16bf bv1 = *(const v16bf*)(bptr + 512);
    acc0 = __builtin_amdgcn_wmma_f32_16x16x32_bf16(
        false, av, false, bv0, (short)0, acc0, false, false);
    acc1 = __builtin_amdgcn_wmma_f32_16x16x32_bf16(
        false, av, false, bv1, (short)0, acc1, false, false);
    aptr += 32;
    bptr += bstride;
  }

  const int col0 = nt0 * 16 + (lane & 15);
  const int col1 = col0 + 16;
  const int rbase = mt * 16 + ((lane < 16) ? 0 : 8);
  const float bs0 = (col0 < ncols) ? bias[col0] : 0.f;
  const float bs1 = (col1 < ncols) ? bias[col1] : 0.f;
#pragma unroll
  for (int r = 0; r < 8; ++r) {
    const int row = rbase + r;
    if (row < M) {
      float* crow = C + (size_t)row * ldc;
      if (col0 < ncols) crow[col0] = acc0[r] + bs0;
      if (col1 < ncols) crow[col1] = acc1[r] + bs1;
    }
  }
}

// ---------------------------------------------------------------------------
// BatchNorm: two-pass batch stats. Coalesced column accumulation + atomics.
// ---------------------------------------------------------------------------
__global__ void bn_stats(const float* __restrict__ X, float* __restrict__ sums,
                         int N, int C, int rows_per) {
  int c = threadIdx.x;
  if (c >= C) return;
  int r0 = blockIdx.x * rows_per;
  int r1 = r0 + rows_per; if (r1 > N) r1 = N;
  float s = 0.f, s2 = 0.f;
  for (int r = r0; r < r1; ++r) {
    float v = X[(size_t)r * C + c];
    s += v; s2 += v * v;
  }
  atomicAdd(&sums[c], s);
  atomicAdd(&sums[C + c], s2);
}

__global__ void bn_apply(const float* __restrict__ X, const float* __restrict__ sums,
                         const float* __restrict__ gamma, const float* __restrict__ beta,
                         float* __restrict__ outF, __bf16* __restrict__ outB,
                         int N, int C, int Cpad, int do_relu) {
  long idx = (long)blockIdx.x * blockDim.x + threadIdx.x;
  if (idx >= (long)N * Cpad) return;
  int n = (int)(idx / Cpad);
  int c = (int)(idx % Cpad);
  if (c < C) {
    float invn = 1.f / (float)N;
    float mean = sums[c] * invn;
    float var  = sums[C + c] * invn - mean * mean;  // biased, like jnp.var
    float y = (X[(size_t)n * C + c] - mean) * rsqrtf(var + 1e-5f) * gamma[c] + beta[c];
    if (do_relu) y = fmaxf(y, 0.f);
    if (outF) outF[(size_t)n * C + c] = y;
    if (outB) outB[idx] = to_bf16(y);
  } else if (outB) {
    outB[idx] = to_bf16(0.f);
  }
}

// out[batch[n]] += h[n], float4-wide
__global__ void readout(const float* __restrict__ h, const int* __restrict__ batch,
                        float* __restrict__ out, int N) {
  const int C4 = EMB / 4;
  long idx = (long)blockIdx.x * blockDim.x + threadIdx.x;
  if (idx >= (long)N * C4) return;
  int n = (int)(idx / C4);
  int c = (int)(idx % C4);
  int g = batch[n];
  float4 v = ((const float4*)(h + (size_t)n * EMB))[c];
  float* op = out + (size_t)g * EMB + (size_t)c * 4;
  atomicAdd(op + 0, v.x);
  atomicAdd(op + 1, v.y);
  atomicAdd(op + 2, v.z);
  atomicAdd(op + 3, v.w);
}

static inline long cdiv(long a, long b) { return (a + b - 1) / b; }

extern "C" void kernel_launch(void* const* d_in, const int* in_sizes, int n_in,
                              void* d_out, int out_size, void* d_ws, size_t ws_size,
                              hipStream_t stream) {
  const int*   x        = (const int*)d_in[0];
  const int*   ei       = (const int*)d_in[1];
  const int*   eat      = (const int*)d_in[2];
  const int*   batch    = (const int*)d_in[3];
  const float* atom_emb = (const float*)d_in[4];
  const float* bond_emb = (const float*)d_in[5];
  const float* W1  = (const float*)d_in[6];
  const float* b1  = (const float*)d_in[7];
  const float* g1  = (const float*)d_in[8];
  const float* bt1 = (const float*)d_in[9];
  const float* W2  = (const float*)d_in[10];
  const float* b2  = (const float*)d_in[11];
  const float* g2  = (const float*)d_in[12];
  const float* bt2 = (const float*)d_in[13];
  float* out = (float*)d_out;

  const int N = in_sizes[0] / 9;
  const int E = in_sizes[1] / 2;
  const int G = out_size / EMB;
  const int* src = ei;
  const int* dst = ei + E;

  // ---- carve workspace (256B aligned chunks) ----
  char* p = (char*)d_ws;
  auto carve = [&](size_t bytes) -> char* {
    char* r = p; p += (bytes + 255) & ~(size_t)255; return r;
  };
  float*  h    = (float*) carve((size_t)N * EMB * 4);
  float*  ea   = (float*) carve((size_t)E * EMB * 4);
  float*  agg  = (float*) carve((size_t)N * EMB * 4);
  float*  z1   = (float*) carve((size_t)N * 600 * 4);
  float*  h2   = (float*) carve((size_t)N * EMB * 4);
  __bf16* a1   = (__bf16*)carve((size_t)N * 320 * 2);
  __bf16* a2   = (__bf16*)carve((size_t)N * 608 * 2);
  __bf16* W1f  = (__bf16*)carve((size_t)LAYERS * 10 * 38 * 512 * 2);
  __bf16* W2f  = (__bf16*)carve((size_t)LAYERS * 19 * 20 * 512 * 2);
  float*  stats= (float*) carve(2 * 608 * 4);
  (void)ws_size; (void)n_in;

  const int BT = 256;

  // Weight fragment prep (tiny; L2-resident afterwards)
  { long t = (long)LAYERS * 10 * 38 * 512;
    prep_w_frag<<<cdiv(t, BT), BT, 0, stream>>>(W1, W1f, 300, 600, 10, 38); }
  { long t = (long)LAYERS * 19 * 20 * 512;
    prep_w_frag<<<cdiv(t, BT), BT, 0, stream>>>(W2, W2f, 600, 300, 19, 20); }

  // Encoders
  { long t = (long)N * EMB; atom_encode<<<cdiv(t, BT), BT, 0, stream>>>(x, atom_emb, h, N); }
  { long t = (long)E * EMB; bond_encode<<<cdiv(t, BT), BT, 0, stream>>>(eat, bond_emb, ea, E); }

  for (int l = 0; l < LAYERS; ++l) {
    // aggregate messages
    hipMemsetAsync(agg, 0, (size_t)N * EMB * 4, stream);
    { long t = (long)E * (EMB / 4);
      edge_msg<<<cdiv(t, BT), BT, 0, stream>>>(h, ea, src, dst, agg, E); }

    // z = h + agg -> bf16 [N,320]
    { long t = (long)N * 320; make_a1<<<cdiv(t, BT), BT, 0, stream>>>(h, agg, a1, N); }

    // GEMM1: [N,320]x[320,608] -> z1 [N,600] (+b1); 19 n-groups of 32 cols
    { long grps = cdiv(N, 16) * (38 / 2);
      gemm_bf16_wmma<<<cdiv(grps, 8), BT, 0, stream>>>(
          a1, 320, W1f + (size_t)l * 10 * 38 * 512, b1 + (size_t)l * 600,
          z1, 600, N, 10, 38, 600); }

    // BN1 + relu -> bf16 [N,608]
    hipMemsetAsync(stats, 0, 2 * 608 * 4, stream);
    { int rp = (N + 239) / 240;
      bn_stats<<<240, 608, 0, stream>>>(z1, stats, N, 600, rp); }
    { long t = (long)N * 608;
      bn_apply<<<cdiv(t, BT), BT, 0, stream>>>(z1, stats, g1 + (size_t)l * 600,
          bt1 + (size_t)l * 600, nullptr, a2, N, 600, 608, 1); }

    // GEMM2: [N,608]x[608,320] -> h2 [N,300] (+b2); 10 n-groups of 32 cols
    { long grps = cdiv(N, 16) * (20 / 2);
      gemm_bf16_wmma<<<cdiv(grps, 8), BT, 0, stream>>>(
          a2, 608, W2f + (size_t)l * 19 * 20 * 512, b2 + (size_t)l * 300,
          h2, 300, N, 19, 20, 300); }

    // BN2 (+relu except last) -> h f32 [N,300]
    hipMemsetAsync(stats, 0, 2 * 608 * 4, stream);
    { int rp = (N + 239) / 240;
      bn_stats<<<240, 320, 0, stream>>>(h2, stats, N, 300, rp); }
    { long t = (long)N * 300;
      bn_apply<<<cdiv(t, BT), BT, 0, stream>>>(h2, stats, g2 + (size_t)l * 300,
          bt2 + (size_t)l * 300, h, nullptr, N, 300, 300, (l < LAYERS - 1) ? 1 : 0); }
  }

  // Graph readout
  hipMemsetAsync(out, 0, (size_t)G * EMB * 4, stream);
  { long t = (long)N * (EMB / 4);
    readout<<<cdiv(t, BT), BT, 0, stream>>>(h, batch, out, N); }
}
